// Decoder_74019466379496
// MI455X (gfx1250) — matrix operand order repaired and checked by reading, not compiled, not run on hardware
//
#include <hip/hip_runtime.h>
#include <hip/hip_bf16.h>
#include <cstdint>
#include <cstddef>

// ---------------------------------------------------------------------------
// Problem constants (from the reference)
// ---------------------------------------------------------------------------
constexpr int Nb  = 64;            // batch
constexpr int ENC = 2048;
constexpr int DEC = 512;
constexpr int ATT = 512;
constexpr int Vv  = 10000;
constexpr int VvP = 10016;         // W_fc packed-B padded to even tile count
constexpr int Pp  = 196;
constexpr int Tt  = 30;
constexpr int Ee  = 128;
constexpr int NP  = Nb * Pp;       // 12544 attention rows
constexpr int KX  = Ee + ENC + DEC;// 2688 = LSTM-input K ([emb, awe, h])
constexpr int G4  = 4 * DEC;       // 2048 gate width

typedef __bf16 bf16_t;
typedef __bf16 v16bf __attribute__((ext_vector_type(16)));
typedef float  v8f   __attribute__((ext_vector_type(8)));

#define DEVFN __device__ __forceinline__

// ---------------------------------------------------------------------------
// WMMA fragment packing helpers (layouts per CDNA5 ISA §7.12.2)
//
// A (16x32 bf16): lane = (m&15) + 16*half ; element j holds
//                 K = j + (j>=8 ? 8 : 0) + 8*half
// B (32x16 bf16): lane = (n&15) + 16*half ; element j holds
//                 K = j + (j>=8 ? 8 : 0) + 8*half
// Packed storage: [tile][kChunk][lane][16 elems] -> each lane's fragment slice
// is one contiguous 32B block => two global_load_b128 per fragment.
// ---------------------------------------------------------------------------
DEVFN void packA_store(bf16_t* dst, int KC, int m, int k, float v) {
    int kc = k >> 5, r = k & 31, half, j;
    if (r < 16) { j = r & 7;              half = r >> 3;        }
    else        { j = ((r - 16) & 7) + 8; half = (r - 16) >> 3; }
    int lane = (m & 15) + (half << 4);
    dst[((((size_t)(m >> 4)) * KC + kc) * 32 + lane) * 16 + j] = (bf16_t)v;
}

DEVFN void packB_store(bf16_t* dst, int KC, int k, int n, float v) {
    int kc = k >> 5, r = k & 31;
    int half = (r >> 3) & 1, j = (r & 7) + ((r >> 4) << 3);
    int lane = (n & 15) + (half << 4);
    dst[((((size_t)(n >> 4)) * KC + kc) * 32 + lane) * 16 + j] = (bf16_t)v;
}

DEVFN float sigmoidf(float x) { return 1.0f / (1.0f + __expf(-x)); }

// ---------------------------------------------------------------------------
// Core GEMM: C[M,N] = A[M,K] @ B[K,N] + bias  (A,B pre-packed bf16 fragments)
// One wave32 computes a 64x32 C block (4 M-tiles x 2 N-tiles): per 32-wide
// K chunk, 6 fragment loads (4 A + 2 B) feed 8 v_wmma_f32_16x16x32_bf16 on
// independent accumulators (no D->A/B hazards, deep XDL pipelining).
// Requires tilesM % 4 == 0 and tilesNpad % 2 == 0 (B packed over tilesNpad
// tiles); stores/bias guarded with n < Ntrue for the padded W_fc case.
// ---------------------------------------------------------------------------
__global__ void __launch_bounds__(256)
gemm_bf16_wmma(const bf16_t* __restrict__ Ap, const bf16_t* __restrict__ Bp,
               const float* __restrict__ bias, float* __restrict__ C,
               int groupsM, int groupsN, int KC, int ldc, int Ntrue)
{
    int wave = blockIdx.x * (blockDim.x >> 5) + (threadIdx.x >> 5);
    if (wave >= groupsM * groupsN) return;
    int gm   = wave / groupsN;
    int gn   = wave - gm * groupsN;
    int lane = threadIdx.x & 31;

    const size_t tileStride = (size_t)KC * 32;   // v16bf elements per tile
    const v16bf* __restrict__ Af = (const v16bf*)Ap + ((size_t)(gm * 4) * tileStride + lane);
    const v16bf* __restrict__ Bf = (const v16bf*)Bp + ((size_t)(gn * 2) * tileStride + lane);

    v8f acc00 = {}, acc01 = {}, acc10 = {}, acc11 = {};
    v8f acc20 = {}, acc21 = {}, acc30 = {}, acc31 = {};
    for (int kc = 0; kc < KC; ++kc) {
        v16bf b0 = Bf[(size_t)kc * 32];
        v16bf b1 = Bf[tileStride     + (size_t)kc * 32];
        v16bf a0 = Af[(size_t)kc * 32];
        v16bf a1 = Af[tileStride     + (size_t)kc * 32];
        v16bf a2 = Af[2 * tileStride + (size_t)kc * 32];
        v16bf a3 = Af[3 * tileStride + (size_t)kc * 32];
        if (kc + 1 < KC) {  // global_prefetch_b8 for next chunk's fragments
            __builtin_prefetch((const void*)(Bf + (size_t)(kc + 1) * 32), 0, 1);
            __builtin_prefetch((const void*)(Bf + tileStride     + (size_t)(kc + 1) * 32), 0, 1);
            __builtin_prefetch((const void*)(Af + (size_t)(kc + 1) * 32), 0, 1);
            __builtin_prefetch((const void*)(Af + tileStride     + (size_t)(kc + 1) * 32), 0, 1);
            __builtin_prefetch((const void*)(Af + 2 * tileStride + (size_t)(kc + 1) * 32), 0, 1);
            __builtin_prefetch((const void*)(Af + 3 * tileStride + (size_t)(kc + 1) * 32), 0, 1);
        }
        acc00 = __builtin_amdgcn_wmma_f32_16x16x32_bf16(false, a0, false, b0, (short)0, acc00, false, false);
        acc10 = __builtin_amdgcn_wmma_f32_16x16x32_bf16(false, a1, false, b0, (short)0, acc10, false, false);
        acc20 = __builtin_amdgcn_wmma_f32_16x16x32_bf16(false, a2, false, b0, (short)0, acc20, false, false);
        acc30 = __builtin_amdgcn_wmma_f32_16x16x32_bf16(false, a3, false, b0, (short)0, acc30, false, false);
        acc01 = __builtin_amdgcn_wmma_f32_16x16x32_bf16(false, a0, false, b1, (short)0, acc01, false, false);
        acc11 = __builtin_amdgcn_wmma_f32_16x16x32_bf16(false, a1, false, b1, (short)0, acc11, false, false);
        acc21 = __builtin_amdgcn_wmma_f32_16x16x32_bf16(false, a2, false, b1, (short)0, acc21, false, false);
        acc31 = __builtin_amdgcn_wmma_f32_16x16x32_bf16(false, a3, false, b1, (short)0, acc31, false, false);
    }

    int half = lane >> 4;
    int col  = lane & 15;
    v8f accs[4][2] = { {acc00, acc01}, {acc10, acc11}, {acc20, acc21}, {acc30, acc31} };
#pragma unroll
    for (int nt = 0; nt < 2; ++nt) {
        int n = (gn * 2 + nt) * 16 + col;
        if (n >= Ntrue) continue;            // padded-N guard (W_fc)
        float bv = bias ? bias[n] : 0.0f;
#pragma unroll
        for (int tt = 0; tt < 4; ++tt) {
            float* __restrict__ Crow =
                C + (size_t)((gm * 4 + tt) * 16 + half * 8) * ldc + n;
#pragma unroll
            for (int r = 0; r < 8; ++r)
                Crow[(size_t)r * ldc] = accs[tt][nt][r] + bv;
        }
    }
}

// ---------------------------------------------------------------------------
// Packing / prep kernels
// ---------------------------------------------------------------------------
// Pack fp32 weight W[Ksrc,Nsrc] into packed-B buffer of logical size
// [Ktot, Ntot] at (rowOff, colOff).
__global__ void pack_b_kernel(const float* __restrict__ W, bf16_t* __restrict__ dst,
                              int Ksrc, int Nsrc, int Ktot, int rowOff, int colOff)
{
    int idx = blockIdx.x * blockDim.x + threadIdx.x;
    if (idx >= Ksrc * Nsrc) return;
    int k = idx / Nsrc, n = idx - k * Nsrc;
    packB_store(dst, Ktot >> 5, k + rowOff, n + colOff, W[(size_t)k * Nsrc + n]);
}

// Zero-fill a bf16 packed buffer (for padded W_fc columns)
__global__ void zerob_kernel(bf16_t* p, size_t n) {
    size_t i = (size_t)blockIdx.x * blockDim.x + threadIdx.x;
    if (i < n) p[i] = (bf16_t)0.0f;
}

// Pack enc rows: A row m = n*P + p, col k = e ; source encoded_image is (N,ENC,P).
__global__ void pack_enc_kernel(const float* __restrict__ img, bf16_t* __restrict__ dst)
{
    size_t idx = (size_t)blockIdx.x * blockDim.x + threadIdx.x;
    if (idx >= (size_t)NP * ENC) return;
    int m = (int)(idx / ENC), k = (int)(idx - (size_t)m * ENC);
    int n = m / Pp, p = m - n * Pp;
    float v = img[((size_t)n * ENC + k) * Pp + p];
    packA_store(dst, ENC >> 5, m, k, v);
}

// mean over pixels + pack as A fragment (M=Nb, K=ENC)
__global__ void mean_pack_kernel(const float* __restrict__ img, bf16_t* __restrict__ dst)
{
    int idx = blockIdx.x * blockDim.x + threadIdx.x;
    if (idx >= Nb * ENC) return;
    int m = idx >> 11, k = idx & (ENC - 1);
    const float* row = img + ((size_t)m * ENC + k) * Pp;
    float s = 0.f;
    for (int p = 0; p < Pp; ++p) s += row[p];
    packA_store(dst, ENC >> 5, m, k, s * (1.0f / Pp));
}

__global__ void zerof_kernel(float* p, int n) {
    int i = blockIdx.x * blockDim.x + threadIdx.x;
    if (i < n) p[i] = 0.0f;
}
__global__ void copyf_kernel(float* dst, const float* src, int n) {
    int i = blockIdx.x * blockDim.x + threadIdx.x;
    if (i < n) dst[i] = src[i];
}
__global__ void addf_kernel(float* dst, const float* a, const float* b, int n) {
    int i = blockIdx.x * blockDim.x + threadIdx.x;
    if (i < n) dst[i] = a[i] + b[i];
}

// h0/c0 -> pack h into hA and xA(h-section), copy c state
__global__ void init_state_kernel(const float* __restrict__ h0c0, float* __restrict__ c,
                                  bf16_t* __restrict__ hA, bf16_t* __restrict__ xA)
{
    int idx = blockIdx.x * blockDim.x + threadIdx.x;
    if (idx >= Nb * DEC) return;
    int m = idx >> 9, d = idx & (DEC - 1);
    float h  = h0c0[(size_t)m * 1024 + d];
    float cv = h0c0[(size_t)m * 1024 + DEC + d];
    c[idx] = cv;
    packA_store(hA, DEC >> 5, m, d, h);
    packA_store(xA, KX  >> 5, m, Ee + ENC + d, h);
}

// ---------------------------------------------------------------------------
// Per-step attention: energies + softmax over P; accumulates alpha_sum.
// One block per batch row; wave32 shuffle reductions.
// ---------------------------------------------------------------------------
__global__ void __launch_bounds__(256)
attention_kernel(const float* __restrict__ att1, const float* __restrict__ att2gate,
                 const float* __restrict__ Wfull, const float* __restrict__ bfull,
                 float* __restrict__ alpha, float* __restrict__ alpha_sum)
{
    __shared__ float satt2[ATT];
    __shared__ float swf[ATT];
    __shared__ float se[Pp];
    __shared__ float sred[8];
    __shared__ float sbc;

    int n = blockIdx.x, tid = threadIdx.x;
    int wv = tid >> 5, lane = tid & 31;

    for (int a = tid; a < ATT; a += 256) {
        satt2[a] = att2gate[(size_t)n * (ATT + ENC) + a];
        swf[a]   = Wfull[a];
    }
    __syncthreads();

    for (int p = wv; p < Pp; p += 8) {
        const float* row = att1 + ((size_t)(n * Pp + p)) * ATT;
        float s = 0.f;
        for (int a = lane; a < ATT; a += 32) {
            float v = row[a] + satt2[a];
            s += (v > 0.f ? v : 0.f) * swf[a];
        }
        for (int off = 16; off; off >>= 1) s += __shfl_xor(s, off, 32);
        if (lane == 0) se[p] = s + bfull[0];
    }
    __syncthreads();

    // softmax
    float mx = -3.4e38f;
    for (int p = tid; p < Pp; p += 256) mx = fmaxf(mx, se[p]);
    for (int off = 16; off; off >>= 1) mx = fmaxf(mx, __shfl_xor(mx, off, 32));
    if (lane == 0) sred[wv] = mx;
    __syncthreads();
    if (tid == 0) {
        float m2 = sred[0];
        for (int i = 1; i < 8; ++i) m2 = fmaxf(m2, sred[i]);
        sbc = m2;
    }
    __syncthreads();
    mx = sbc;

    float ss = 0.f;
    for (int p = tid; p < Pp; p += 256) { float ev = __expf(se[p] - mx); se[p] = ev; ss += ev; }
    for (int off = 16; off; off >>= 1) ss += __shfl_xor(ss, off, 32);
    __syncthreads();
    if (lane == 0) sred[wv] = ss;
    __syncthreads();
    if (tid == 0) {
        float s2 = 0.f;
        for (int i = 0; i < 8; ++i) s2 += sred[i];
        sbc = 1.0f / s2;
    }
    __syncthreads();
    float inv = sbc;
    for (int p = tid; p < Pp; p += 256) {
        float a = se[p] * inv;
        alpha[(size_t)n * Pp + p] = a;
        alpha_sum[(size_t)n * Pp + p] += a;
    }
}

// ---------------------------------------------------------------------------
// awe = gate * (alpha-weighted enc); writes x = [emb_t, awe] (packed bf16 A)
// ---------------------------------------------------------------------------
__global__ void awe_pack_kernel(const float* __restrict__ img, const float* __restrict__ alpha,
                                const float* __restrict__ att2gate,
                                const float* __restrict__ embeddings, int t,
                                bf16_t* __restrict__ xA)
{
    int idx = blockIdx.x * blockDim.x + threadIdx.x;
    if (idx >= Nb * (Ee + ENC)) return;
    int m = idx / (Ee + ENC), k = idx - m * (Ee + ENC);
    float v;
    if (k < Ee) {
        v = embeddings[((size_t)m * Tt + t) * Ee + k];
    } else {
        int e = k - Ee;
        const float* ip = img + ((size_t)m * ENC + e) * Pp;
        const float* ap = alpha + (size_t)m * Pp;
        float s = 0.f;
        for (int p = 0; p < Pp; ++p) s += ip[p] * ap[p];
        v = att2gate[(size_t)m * (ATT + ENC) + ATT + e] * s;   // 'gate' linear (no sigmoid, faithful)
    }
    packA_store(xA, KX >> 5, m, k, v);
}

// ---------------------------------------------------------------------------
// LSTM pointwise: gates (i,f,g,o order) -> new c ; h packed into hA and xA
// ---------------------------------------------------------------------------
__global__ void lstm_kernel(const float* __restrict__ gates, float* __restrict__ c,
                            bf16_t* __restrict__ hA, bf16_t* __restrict__ xA)
{
    int idx = blockIdx.x * blockDim.x + threadIdx.x;
    if (idx >= Nb * DEC) return;
    int m = idx >> 9, d = idx & (DEC - 1);
    const float* g = gates + (size_t)m * G4;
    float gi = g[d], gf = g[DEC + d], gg = g[2 * DEC + d], go = g[3 * DEC + d];
    float cn = sigmoidf(gf) * c[idx] + sigmoidf(gi) * tanhf(gg);
    float hn = sigmoidf(go) * tanhf(cn);
    c[idx] = cn;
    packA_store(hA, DEC >> 5, m, d, hn);
    packA_store(xA, KX  >> 5, m, Ee + ENC + d, hn);
}

// loss = ALPHA_C * mean((1 - sum_t alpha)^2)
__global__ void loss_kernel(const float* __restrict__ alpha_sum, float* __restrict__ out)
{
    __shared__ float red[256];
    int tid = threadIdx.x;
    float s = 0.f;
    for (int i = tid; i < NP; i += 256) {
        float d = 1.0f - alpha_sum[i];
        s += d * d;
    }
    red[tid] = s; __syncthreads();
    for (int off = 128; off; off >>= 1) {
        if (tid < off) red[tid] += red[tid + off];
        __syncthreads();
    }
    if (tid == 0) out[0] = 1.01f * red[0] / (float)NP;
}

// ---------------------------------------------------------------------------
// Workspace layout
// ---------------------------------------------------------------------------
constexpr size_t a256(size_t x) { return (x + 255) & ~(size_t)255; }
constexpr size_t OFF_ENCA  = 0;
constexpr size_t OFF_ATT1  = a256(OFF_ENCA  + (size_t)NP * ENC * 2);
constexpr size_t OFF_WEB   = a256(OFF_ATT1  + (size_t)NP * ATT * 4);
constexpr size_t OFF_WDG   = a256(OFF_WEB   + (size_t)ENC * ATT * 2);
constexpr size_t OFF_WIHH  = a256(OFF_WDG   + (size_t)DEC * (ATT + ENC) * 2);
constexpr size_t OFF_WFC   = a256(OFF_WIHH  + (size_t)KX * G4 * 2);
constexpr size_t OFF_WINIT = a256(OFF_WFC   + (size_t)DEC * VvP * 2);
constexpr size_t OFF_MEANA = a256(OFF_WINIT + (size_t)ENC * 1024 * 2);
constexpr size_t OFF_H0C0  = a256(OFF_MEANA + (size_t)Nb * ENC * 2);
constexpr size_t OFF_A2G   = a256(OFF_H0C0  + (size_t)Nb * 1024 * 4);
constexpr size_t OFF_ALPHA = a256(OFF_A2G   + (size_t)Nb * (ATT + ENC) * 4);
constexpr size_t OFF_ASUM  = a256(OFF_ALPHA + (size_t)NP * 4);
constexpr size_t OFF_XA    = a256(OFF_ASUM  + (size_t)NP * 4);
constexpr size_t OFF_HA    = a256(OFF_XA    + (size_t)Nb * KX * 2);
constexpr size_t OFF_C     = a256(OFF_HA    + (size_t)Nb * DEC * 2);
constexpr size_t OFF_GATES = a256(OFF_C     + (size_t)Nb * DEC * 4);
constexpr size_t OFF_BIHH  = a256(OFF_GATES + (size_t)Nb * G4 * 4);
constexpr size_t OFF_BDG   = a256(OFF_BIHH  + (size_t)G4 * 4);
constexpr size_t OFF_BINIT = a256(OFF_BDG   + (size_t)(ATT + ENC) * 4);

// ---------------------------------------------------------------------------
extern "C" void kernel_launch(void* const* d_in, const int* in_sizes, int n_in,
                              void* d_out, int out_size, void* d_ws, size_t ws_size,
                              hipStream_t stream)
{
    const float* img       = (const float*)d_in[0];   // (N,ENC,P)
    const float* emb       = (const float*)d_in[1];   // (N,T,E)
    const float* W_init_h  = (const float*)d_in[5];
    const float* b_init_h  = (const float*)d_in[6];
    const float* W_init_c  = (const float*)d_in[7];
    const float* b_init_c  = (const float*)d_in[8];
    const float* W_enc_att = (const float*)d_in[9];
    const float* b_enc_att = (const float*)d_in[10];
    const float* W_dec_att = (const float*)d_in[11];
    const float* b_dec_att = (const float*)d_in[12];
    const float* W_full    = (const float*)d_in[13];
    const float* b_full    = (const float*)d_in[14];
    const float* W_gate    = (const float*)d_in[15];
    const float* b_gate    = (const float*)d_in[16];
    const float* W_ih      = (const float*)d_in[17];
    const float* b_ih      = (const float*)d_in[18];
    const float* W_hh      = (const float*)d_in[19];
    const float* b_hh      = (const float*)d_in[20];
    const float* W_fc      = (const float*)d_in[21];
    const float* b_fc      = (const float*)d_in[22];

    char* ws = (char*)d_ws;
    bf16_t* encA   = (bf16_t*)(ws + OFF_ENCA);
    float*  att1   = (float*) (ws + OFF_ATT1);
    bf16_t* WencB  = (bf16_t*)(ws + OFF_WEB);
    bf16_t* WdgB   = (bf16_t*)(ws + OFF_WDG);
    bf16_t* WihhB  = (bf16_t*)(ws + OFF_WIHH);
    bf16_t* WfcB   = (bf16_t*)(ws + OFF_WFC);
    bf16_t* WinitB = (bf16_t*)(ws + OFF_WINIT);
    bf16_t* meanA  = (bf16_t*)(ws + OFF_MEANA);
    float*  h0c0   = (float*) (ws + OFF_H0C0);
    float*  a2g    = (float*) (ws + OFF_A2G);
    float*  alpha  = (float*) (ws + OFF_ALPHA);
    float*  asum   = (float*) (ws + OFF_ASUM);
    bf16_t* xA     = (bf16_t*)(ws + OFF_XA);
    bf16_t* hA     = (bf16_t*)(ws + OFF_HA);
    float*  cst    = (float*) (ws + OFF_C);
    float*  gates  = (float*) (ws + OFF_GATES);
    float*  bihh   = (float*) (ws + OFF_BIHH);
    float*  bdg    = (float*) (ws + OFF_BDG);
    float*  binit  = (float*) (ws + OFF_BINIT);

    float* preds = (float*)d_out;            // (N, V)
    float* loss  = (float*)d_out + Nb * Vv;  // scalar

    auto blk = [](size_t n) { return dim3((unsigned)((n + 255) / 256)); };
    // Npad: packed-B logical width (multiple of 32 columns); Ntrue: real width
    auto gemm = [&](const bf16_t* A, const bf16_t* B, const float* bias, float* C,
                    int M, int Npad, int Ntrue, int K, int ldc) {
        int groupsM = M >> 6;            // 4 M-tiles (64 rows) per wave
        int groupsN = Npad >> 5;         // 2 N-tiles (32 cols) per wave
        int KC = K >> 5;
        int waves = groupsM * groupsN;
        hipLaunchKernelGGL(gemm_bf16_wmma, dim3((waves + 7) / 8), dim3(256), 0, stream,
                           A, B, bias, C, groupsM, groupsN, KC, ldc, Ntrue);
    };

    // ---------------- Prep (time-invariant) ----------------
    hipLaunchKernelGGL(pack_b_kernel, blk((size_t)ENC * DEC), dim3(256), 0, stream,
                       W_init_h, WinitB, ENC, DEC, ENC, 0, 0);
    hipLaunchKernelGGL(pack_b_kernel, blk((size_t)ENC * DEC), dim3(256), 0, stream,
                       W_init_c, WinitB, ENC, DEC, ENC, 0, DEC);
    hipLaunchKernelGGL(pack_b_kernel, blk((size_t)ENC * ATT), dim3(256), 0, stream,
                       W_enc_att, WencB, ENC, ATT, ENC, 0, 0);
    hipLaunchKernelGGL(pack_b_kernel, blk((size_t)DEC * ATT), dim3(256), 0, stream,
                       W_dec_att, WdgB, DEC, ATT, DEC, 0, 0);
    hipLaunchKernelGGL(pack_b_kernel, blk((size_t)DEC * ENC), dim3(256), 0, stream,
                       W_gate, WdgB, DEC, ENC, DEC, 0, ATT);
    hipLaunchKernelGGL(pack_b_kernel, blk((size_t)(Ee + ENC) * G4), dim3(256), 0, stream,
                       W_ih, WihhB, Ee + ENC, G4, KX, 0, 0);
    hipLaunchKernelGGL(pack_b_kernel, blk((size_t)DEC * G4), dim3(256), 0, stream,
                       W_hh, WihhB, DEC, G4, KX, Ee + ENC, 0);
    hipLaunchKernelGGL(zerob_kernel, blk((size_t)DEC * VvP), dim3(256), 0, stream,
                       WfcB, (size_t)DEC * VvP);
    hipLaunchKernelGGL(pack_b_kernel, blk((size_t)DEC * Vv), dim3(256), 0, stream,
                       W_fc, WfcB, DEC, Vv, DEC, 0, 0);
    hipLaunchKernelGGL(pack_enc_kernel, blk((size_t)NP * ENC), dim3(256), 0, stream, img, encA);
    hipLaunchKernelGGL(mean_pack_kernel, blk((size_t)Nb * ENC), dim3(256), 0, stream, img, meanA);
    hipLaunchKernelGGL(zerof_kernel, blk(NP), dim3(256), 0, stream, asum, NP);
    hipLaunchKernelGGL(addf_kernel, blk(G4), dim3(256), 0, stream, bihh, b_ih, b_hh, G4);
    hipLaunchKernelGGL(copyf_kernel, blk(ATT), dim3(256), 0, stream, bdg, b_dec_att, ATT);
    hipLaunchKernelGGL(copyf_kernel, blk(ENC), dim3(256), 0, stream, bdg + ATT, b_gate, ENC);
    hipLaunchKernelGGL(copyf_kernel, blk(DEC), dim3(256), 0, stream, binit, b_init_h, DEC);
    hipLaunchKernelGGL(copyf_kernel, blk(DEC), dim3(256), 0, stream, binit + DEC, b_init_c, DEC);

    // h0|c0 = mean_enc @ [W_init_h | W_init_c] + [b_h | b_c]
    gemm(meanA, WinitB, binit, h0c0, Nb, 1024, 1024, ENC, 1024);
    hipLaunchKernelGGL(init_state_kernel, blk((size_t)Nb * DEC), dim3(256), 0, stream,
                       h0c0, cst, hA, xA);

    // att1 = enc @ W_enc_att + b   (biggest GEMM: 12544 x 512 x 2048)
    gemm(encA, WencB, b_enc_att, att1, NP, ATT, ATT, ENC, ATT);

    // ---------------- Time loop ----------------
    for (int t = 0; t < Tt; ++t) {
        // [att2 | gate] = h @ [W_dec_att | W_gate] + [b_dec_att | b_gate]
        gemm(hA, WdgB, bdg, a2g, Nb, ATT + ENC, ATT + ENC, DEC, ATT + ENC);
        hipLaunchKernelGGL(attention_kernel, dim3(Nb), dim3(256), 0, stream,
                           att1, a2g, W_full, b_full, alpha, asum);
        hipLaunchKernelGGL(awe_pack_kernel, blk((size_t)Nb * (Ee + ENC)), dim3(256), 0, stream,
                           img, alpha, a2g, emb, t, xA);
        // gates = [emb, awe, h] @ [W_ih ; W_hh] + (b_ih + b_hh)
        gemm(xA, WihhB, bihh, gates, Nb, G4, G4, KX, G4);
        hipLaunchKernelGGL(lstm_kernel, blk((size_t)Nb * DEC), dim3(256), 0, stream,
                           gates, cst, hA, xA);
    }

    // preds (only final step is returned) + loss
    gemm(hA, WfcB, b_fc, preds, Nb, VvP, Vv, DEC, Vv);
    hipLaunchKernelGGL(loss_kernel, dim3(1), dim3(256), 0, stream, asum, loss);
}
